// TokenGlobalRelation_48524540510925
// MI455X (gfx1250) — matrix operand order repaired and checked by reading, not compiled, not run on hardware
//
#include <hip/hip_runtime.h>
#include <hip/hip_bf16.h>
#include <cstdint>

// ---------------------------------------------------------------------------
// Types / helpers for CDNA5 WMMA (wave32, V_WMMA_F32_16X16X32_F16)
// ---------------------------------------------------------------------------
typedef _Float16 h16;
typedef __attribute__((ext_vector_type(16))) _Float16 v16h;
typedef __attribute__((ext_vector_type(8)))  _Float16 v8h;
typedef __attribute__((ext_vector_type(8)))  float    v8f;

__device__ __forceinline__ v8f vzero8() {
  v8f z = {0.f, 0.f, 0.f, 0.f, 0.f, 0.f, 0.f, 0.f};
  return z;
}

__device__ __forceinline__ float gelu_exact(float x) {
  return 0.5f * x * (1.f + erff(x * 0.70710678118654752f));
}

// Build a 16-element f16 fragment from two 8-halves (each a 16B load).
__device__ __forceinline__ v16h make_frag(const h16* lo, const h16* hi) {
  v8h a = *reinterpret_cast<const v8h*>(lo);
  v8h b = *reinterpret_cast<const v8h*>(hi);
  v16h r;
#pragma unroll
  for (int i = 0; i < 8; ++i) { r[i] = a[i]; r[i + 8] = b[i]; }
  return r;
}

// A-matrix fragment (16x32, row-major source, stride ldx in elements).
// Layout: lane group g holds K = i + 16*g (contiguous 16) for row
// M = lane&15, matching b_frag.
__device__ __forceinline__ v16h a_frag(const h16* X, int ldx, int lane) {
  int g = lane >> 4, m = lane & 15;
  const h16* p = X + m * ldx + 16 * g;
  return make_frag(p, p + 8);
}

// B-matrix fragment (32x16) from a transposed tile BT[n][k] (k contiguous,
// stride ldb elements). Layout used here (matches a_frag): lane group g holds K = i + 16*g for
// column N = lane&15  ->  contiguous 16 f16 per lane.
__device__ __forceinline__ v16h b_frag(const h16* BT, int ldb, int lane) {
  int g = lane >> 4, n = lane & 15;
  const h16* p = BT + n * ldb + 16 * g;
  return make_frag(p, p + 8);
}

__device__ __forceinline__ v8f wmma_f16(v16h a, v16h b, v8f c) {
  return __builtin_amdgcn_wmma_f32_16x16x32_f16(false, a, false, b,
                                                (short)0, c, false, false);
}

// ---------------------------------------------------------------------------
// Staging helpers: global f32 (float4 reads) -> LDS f16 transposed [n][k=40]
// ---------------------------------------------------------------------------
// 32 K x 128 N weight tile
__device__ __forceinline__ void stage_bt128(h16* dst, const float* __restrict__ W,
                                            int ldw, int k0, int n0, int tid) {
  for (int idx = tid; idx < 32 * 32; idx += 256) {
    int k = idx >> 5, n4 = (idx & 31) << 2;
    float4 w4 = *reinterpret_cast<const float4*>(
        &W[(size_t)(k0 + k) * ldw + n0 + n4]);
    dst[(n4 + 0) * 40 + k] = (h16)w4.x;
    dst[(n4 + 1) * 40 + k] = (h16)w4.y;
    dst[(n4 + 2) * 40 + k] = (h16)w4.z;
    dst[(n4 + 3) * 40 + k] = (h16)w4.w;
  }
}

// 32 K x 256 N weight tile
__device__ __forceinline__ void stage_bt256(h16* dst, const float* __restrict__ W,
                                            int ldw, int k0, int n0, int tid) {
  for (int idx = tid; idx < 32 * 64; idx += 256) {
    int k = idx >> 6, n4 = (idx & 63) << 2;
    float4 w4 = *reinterpret_cast<const float4*>(
        &W[(size_t)(k0 + k) * ldw + n0 + n4]);
    dst[(n4 + 0) * 40 + k] = (h16)w4.x;
    dst[(n4 + 1) * 40 + k] = (h16)w4.y;
    dst[(n4 + 2) * 40 + k] = (h16)w4.z;
    dst[(n4 + 3) * 40 + k] = (h16)w4.w;
  }
}

// 32 keys x 256 ch f16 v-tile -> LDS transposed [c][k=40]
__device__ __forceinline__ void stage_vt(h16* dst, const h16* __restrict__ vmat,
                                         int b, int k0, int tid) {
  for (int idx = tid; idx < 32 * 32; idx += 256) {
    int k = idx >> 5, c8 = (idx & 31) << 3;
    v8h v = *reinterpret_cast<const v8h*>(
        &vmat[(size_t)(b * 576 + k0 + k) * 256 + c8]);
#pragma unroll
    for (int i = 0; i < 8; ++i) dst[(c8 + i) * 40 + k] = v[i];
  }
}

// ---------------------------------------------------------------------------
// Kernel 1: gather 576 refer points, depth MLP (dp1/dp2) + global_pos add,
//           then kv MLP (kv1/kv2) -> k, v  (f16).  Grid: (9, B), 256 thr.
// ---------------------------------------------------------------------------
__global__ __launch_bounds__(256) void refer_kv_kernel(
    const float* __restrict__ depth_token, const float* __restrict__ global_pos,
    const float* __restrict__ refer_coords,
    const float* __restrict__ dp1_w, const float* __restrict__ dp1_b,
    const float* __restrict__ dp2_w, const float* __restrict__ dp2_b,
    const float* __restrict__ kv1_w, const float* __restrict__ kv1_b,
    const float* __restrict__ kv2_w, const float* __restrict__ kv2_b,
    h16* __restrict__ kbuf, h16* __restrict__ vbuf) {
  const int b = blockIdx.y;
  const int r0 = blockIdx.x * 64;
  const int tid = threadIdx.x;
  const int lane = tid & 31, wid = tid >> 5;
  const int mblk = wid & 3, nh = wid >> 2;  // 4 row blocks x 2 col halves

  __shared__ int   sPix[64];
  __shared__ float sVal[64];
  __shared__ h16 lA[64 * 40];
  __shared__ h16 lBT[256 * 40];
  __shared__ h16 lH[64 * 264];
  __shared__ h16 lX[64 * 264];

  if (tid < 64) {
    int j = r0 + tid;
    float gx = refer_coords[(size_t)(b * 576 + j) * 2 + 0];
    float gy = refer_coords[(size_t)(b * 576 + j) * 2 + 1];
    float ix = rintf(((gx + 1.f) * 64.f - 1.f) * 0.5f);
    float iy = rintf(((gy + 1.f) * 64.f - 1.f) * 0.5f);
    bool ok = (ix >= 0.f) && (ix < 64.f) && (iy >= 0.f) && (iy < 64.f);
    int ixc = (int)fminf(fmaxf(ix, 0.f), 63.f);
    int iyc = (int)fminf(fmaxf(iy, 0.f), 63.f);
    sPix[tid] = iyc * 64 + ixc;
    sVal[tid] = ok ? 1.f : 0.f;
  }
  __syncthreads();

  v8f acc[8];
  v16h bf[8];

  // ---- depth MLP layer 1 (gathered input, GELU) -> lH
#pragma unroll
  for (int j = 0; j < 8; ++j) acc[j] = vzero8();
  for (int k0 = 0; k0 < 256; k0 += 32) {
    __syncthreads();
    for (int idx = tid; idx < 64 * 32; idx += 256) {
      int k = idx / 64, r = idx % 64;
      lA[r * 40 + k] =
          (h16)depth_token[((size_t)b * 256 + k0 + k) * 4096 + sPix[r]];
    }
    stage_bt256(lBT, dp1_w, 256, k0, 0, tid);
    __syncthreads();
    v16h a = a_frag(&lA[(16 * mblk) * 40], 40, lane);
#pragma unroll
    for (int j = 0; j < 8; ++j)
      bf[j] = b_frag(&lBT[(128 * nh + 16 * j) * 40], 40, lane);
#pragma unroll
    for (int j = 0; j < 8; ++j) acc[j] = wmma_f16(a, bf[j], acc[j]);
  }
  __syncthreads();
#pragma unroll
  for (int j = 0; j < 8; ++j)
#pragma unroll
    for (int r = 0; r < 8; ++r) {
      int row = 16 * mblk + r + 8 * (lane >> 4);
      int col = 128 * nh + 16 * j + (lane & 15);
      lH[row * 264 + col] = (h16)gelu_exact(acc[j][r] + dp1_b[col]);
    }
  __syncthreads();

  // ---- depth MLP layer 2 + global_pos gather + valid mask -> lX
#pragma unroll
  for (int j = 0; j < 8; ++j) acc[j] = vzero8();
  for (int k0 = 0; k0 < 256; k0 += 32) {
    __syncthreads();
    stage_bt256(lBT, dp2_w, 256, k0, 0, tid);
    __syncthreads();
    v16h a = a_frag(&lH[(16 * mblk) * 264 + k0], 264, lane);
#pragma unroll
    for (int j = 0; j < 8; ++j)
      bf[j] = b_frag(&lBT[(128 * nh + 16 * j) * 40], 40, lane);
#pragma unroll
    for (int j = 0; j < 8; ++j) acc[j] = wmma_f16(a, bf[j], acc[j]);
  }
  __syncthreads();
#pragma unroll
  for (int j = 0; j < 8; ++j)
#pragma unroll
    for (int r = 0; r < 8; ++r) {
      int row = 16 * mblk + r + 8 * (lane >> 4);
      int col = 128 * nh + 16 * j + (lane & 15);
      float v = acc[j][r] + dp2_b[col] +
                global_pos[((size_t)b * 256 + col) * 4096 + sPix[row]];
      lX[row * 264 + col] = (h16)(v * sVal[row]);
    }
  __syncthreads();

  // ---- kv layer 1 (GELU) -> lH
#pragma unroll
  for (int j = 0; j < 8; ++j) acc[j] = vzero8();
  for (int k0 = 0; k0 < 256; k0 += 32) {
    __syncthreads();
    stage_bt256(lBT, kv1_w, 256, k0, 0, tid);
    __syncthreads();
    v16h a = a_frag(&lX[(16 * mblk) * 264 + k0], 264, lane);
#pragma unroll
    for (int j = 0; j < 8; ++j)
      bf[j] = b_frag(&lBT[(128 * nh + 16 * j) * 40], 40, lane);
#pragma unroll
    for (int j = 0; j < 8; ++j) acc[j] = wmma_f16(a, bf[j], acc[j]);
  }
  __syncthreads();
#pragma unroll
  for (int j = 0; j < 8; ++j)
#pragma unroll
    for (int r = 0; r < 8; ++r) {
      int row = 16 * mblk + r + 8 * (lane >> 4);
      int col = 128 * nh + 16 * j + (lane & 15);
      lH[row * 264 + col] = (h16)gelu_exact(acc[j][r] + kv1_b[col]);
    }
  __syncthreads();

  // ---- kv layer 2 (N=512, two 256-wide passes) -> kbuf / vbuf
  for (int np = 0; np < 2; ++np) {
#pragma unroll
    for (int j = 0; j < 8; ++j) acc[j] = vzero8();
    for (int k0 = 0; k0 < 256; k0 += 32) {
      __syncthreads();
      stage_bt256(lBT, kv2_w, 512, k0, np * 256, tid);
      __syncthreads();
      v16h a = a_frag(&lH[(16 * mblk) * 264 + k0], 264, lane);
#pragma unroll
      for (int j = 0; j < 8; ++j)
        bf[j] = b_frag(&lBT[(128 * nh + 16 * j) * 40], 40, lane);
#pragma unroll
      for (int j = 0; j < 8; ++j) acc[j] = wmma_f16(a, bf[j], acc[j]);
    }
    h16* dst = np ? vbuf : kbuf;
#pragma unroll
    for (int j = 0; j < 8; ++j)
#pragma unroll
      for (int r = 0; r < 8; ++r) {
        int row = 16 * mblk + r + 8 * (lane >> 4);
        int col = 128 * nh + 16 * j + (lane & 15);
        dst[((size_t)b * 576 + r0 + row) * 256 + col] =
            (h16)(acc[j][r] + kv2_b[np * 256 + col]);
      }
  }
}

// ---------------------------------------------------------------------------
// Kernel 2: seg_rel 2-layer MLP -> seg_gemo[:, 0:256].  Grid: (64, B).
// ---------------------------------------------------------------------------
__global__ __launch_bounds__(256) void segrel_kernel(
    const float* __restrict__ seg_token,
    const float* __restrict__ sp1_w, const float* __restrict__ sp1_b,
    const float* __restrict__ sp2_w, const float* __restrict__ sp2_b,
    h16* __restrict__ gemo) {
  const int b = blockIdx.y;
  const int n0 = blockIdx.x * 64;
  const int tid = threadIdx.x, lane = tid & 31, wid = tid >> 5;
  const int mblk = wid & 3, nh = wid >> 2;
  __shared__ h16 lA[64 * 40];
  __shared__ h16 lBT[256 * 40];
  __shared__ h16 lH[64 * 264];
  v8f acc[8];
  v16h bf[8];

#pragma unroll
  for (int j = 0; j < 8; ++j) acc[j] = vzero8();
  for (int k0 = 0; k0 < 256; k0 += 32) {
    __syncthreads();
    for (int idx = tid; idx < 64 * 32 / 4; idx += 256) {
      int k = idx >> 4, n4 = (idx & 15) << 2;  // transposed, float4 along n
      float4 t = *reinterpret_cast<const float4*>(
          &seg_token[((size_t)b * 256 + k0 + k) * 4096 + n0 + n4]);
      lA[(n4 + 0) * 40 + k] = (h16)t.x;
      lA[(n4 + 1) * 40 + k] = (h16)t.y;
      lA[(n4 + 2) * 40 + k] = (h16)t.z;
      lA[(n4 + 3) * 40 + k] = (h16)t.w;
    }
    stage_bt256(lBT, sp1_w, 256, k0, 0, tid);
    __syncthreads();
    v16h a = a_frag(&lA[(16 * mblk) * 40], 40, lane);
#pragma unroll
    for (int j = 0; j < 8; ++j)
      bf[j] = b_frag(&lBT[(128 * nh + 16 * j) * 40], 40, lane);
#pragma unroll
    for (int j = 0; j < 8; ++j) acc[j] = wmma_f16(a, bf[j], acc[j]);
  }
  __syncthreads();
#pragma unroll
  for (int j = 0; j < 8; ++j)
#pragma unroll
    for (int r = 0; r < 8; ++r) {
      int row = 16 * mblk + r + 8 * (lane >> 4);
      int col = 128 * nh + 16 * j + (lane & 15);
      lH[row * 264 + col] = (h16)gelu_exact(acc[j][r] + sp1_b[col]);
    }
  __syncthreads();

#pragma unroll
  for (int j = 0; j < 8; ++j) acc[j] = vzero8();
  for (int k0 = 0; k0 < 256; k0 += 32) {
    __syncthreads();
    stage_bt256(lBT, sp2_w, 256, k0, 0, tid);
    __syncthreads();
    v16h a = a_frag(&lH[(16 * mblk) * 264 + k0], 264, lane);
#pragma unroll
    for (int j = 0; j < 8; ++j)
      bf[j] = b_frag(&lBT[(128 * nh + 16 * j) * 40], 40, lane);
#pragma unroll
    for (int j = 0; j < 8; ++j) acc[j] = wmma_f16(a, bf[j], acc[j]);
  }
#pragma unroll
  for (int j = 0; j < 8; ++j)
#pragma unroll
    for (int r = 0; r < 8; ++r) {
      int row = 16 * mblk + r + 8 * (lane >> 4);
      int col = 128 * nh + 16 * j + (lane & 15);
      gemo[((size_t)b * 4096 + n0 + row) * 1024 + col] =
          (h16)(acc[j][r] + sp2_b[col]);
    }
}

// ---------------------------------------------------------------------------
// Kernel 3: geometry -> seg_gemo[:, 256:832] (dist), [:, 832:1024] (intri).
// One thread per (b, n).  Grid: 128 blocks x 256.
// ---------------------------------------------------------------------------
__global__ __launch_bounds__(256) void geom_kernel(
    const float* __restrict__ refer_coords, h16* __restrict__ gemo) {
  int t = blockIdx.x * 256 + threadIdx.x;
  int b = t >> 12, n = t & 4095;
  float gx = (float)(n & 63) * (2.f / 63.f) - 1.f;
  float gy = (float)(n >> 6) * (2.f / 63.f) - 1.f;
  float px = (gx + 1.f) * 126.f;  // (g+1)*2*(W-1)
  float py = (gy + 1.f) * 126.f;
  h16* row = gemo + ((size_t)b * 4096 + n) * 1024;
  const float* rc = refer_coords + (size_t)b * 576 * 2;
  for (int p = 0; p < 192; ++p) {
    float ax = rc[p * 6 + 0], ay = rc[p * 6 + 1];
    float bx = rc[p * 6 + 2], by = rc[p * 6 + 3];
    float cx = rc[p * 6 + 4], cy = rc[p * 6 + 5];
    row[256 + p * 3 + 0] =
        (h16)(sqrtf((gx - ax) * (gx - ax) + (gy - ay) * (gy - ay)) * 0.5f);
    row[256 + p * 3 + 1] =
        (h16)(sqrtf((gx - bx) * (gx - bx) + (gy - by) * (gy - by)) * 0.5f);
    row[256 + p * 3 + 2] =
        (h16)(sqrtf((gx - cx) * (gx - cx) + (gy - cy) * (gy - cy)) * 0.5f);
    float Ax = (ax + 1.f) * 126.f, Ay = (ay + 1.f) * 126.f;
    float Bx = (bx + 1.f) * 126.f, By = (by + 1.f) * 126.f;
    float Cx = (cx + 1.f) * 126.f, Cy = (cy + 1.f) * 126.f;
    float s1 = (px - Bx) * (Ay - By) - (Ax - Bx) * (py - By);
    float s2 = (px - Cx) * (By - Cy) - (Bx - Cx) * (py - Cy);
    float s3 = (px - Ax) * (Cy - Ay) - (Cx - Ax) * (py - Ay);
    bool hn = (s1 < 0.f) || (s2 < 0.f) || (s3 < 0.f);
    bool hp = (s1 > 0.f) || (s2 > 0.f) || (s3 > 0.f);
    row[832 + p] = (h16)((hn && hp) ? 0.f : 1.f);
  }
}

// ---------------------------------------------------------------------------
// Generic WMMA GEMM: Y = act(X[f16] @ W[f32] + bias), 64x128 tile / block,
// double-buffered B staging (one barrier per K-step) + prefetch hints.
// OUTMODE 0: f16 row-major; OUTMODE 1: f32 transposed to (B, C, HW).
// ---------------------------------------------------------------------------
template <int ACT, int OUTMODE>
__global__ __launch_bounds__(256) void gemm_x_w(
    const h16* __restrict__ X, int ldx, const float* __restrict__ W, int N,
    const float* __restrict__ bias, int K, h16* __restrict__ Yh, int ldy,
    float* __restrict__ Yt) {
  const int r0 = blockIdx.x * 64;
  const int n0 = blockIdx.y * 128;
  const int tid = threadIdx.x, lane = tid & 31, wid = tid >> 5;
  const int mblk = wid & 3, nq = wid >> 2;  // 4 M-blocks x 2 N-halves(64)
  __shared__ h16 lBT[2][128 * 40];
  v8f acc[4];
#pragma unroll
  for (int j = 0; j < 4; ++j) acc[j] = vzero8();

  stage_bt128(lBT[0], W, N, 0, n0, tid);
  int buf = 0;
  for (int k0 = 0; k0 < K; k0 += 32) {
    __syncthreads();
    if (k0 + 32 < K) {
      if (k0 + 64 < K)  // speculative prefetch of the tile after next
        __builtin_prefetch(&W[(size_t)(k0 + 64) * N + n0], 0, 0);
      stage_bt128(lBT[buf ^ 1], W, N, k0 + 32, n0, tid);
    }
    v16h a = a_frag(X + (size_t)(r0 + 16 * mblk) * ldx + k0, ldx, lane);
    v16h bf[4];
#pragma unroll
    for (int j = 0; j < 4; ++j)
      bf[j] = b_frag(&lBT[buf][(64 * nq + 16 * j) * 40], 40, lane);
#pragma unroll
    for (int j = 0; j < 4; ++j) acc[j] = wmma_f16(a, bf[j], acc[j]);
    buf ^= 1;
  }
#pragma unroll
  for (int j = 0; j < 4; ++j)
#pragma unroll
    for (int r = 0; r < 8; ++r) {
      int row = r0 + 16 * mblk + r + 8 * (lane >> 4);
      int col = n0 + 64 * nq + 16 * j + (lane & 15);
      float v = acc[j][r] + bias[col];
      if (ACT) v = gelu_exact(v);
      if (OUTMODE == 0) {
        Yh[(size_t)row * ldy + col] = (h16)v;
      } else {
        int bb = row >> 12, nn = row & 4095;
        Yt[((size_t)bb * 256 + col) * 4096 + nn] = v;
      }
    }
}

// ---------------------------------------------------------------------------
// Kernel 6: attention over 576 keys. 32 query rows / block, softmax in LDS,
// double-buffered value tiles.  Grid: (128, B), 256 threads.
// ---------------------------------------------------------------------------
__global__ __launch_bounds__(256) void attn_kernel(
    const h16* __restrict__ q, const h16* __restrict__ kmat,
    const h16* __restrict__ vmat, h16* __restrict__ fused) {
  const int b = blockIdx.y;
  const int q0 = blockIdx.x * 32;
  const int tid = threadIdx.x, lane = tid & 31, wid = tid >> 5;
  __shared__ h16 lP[32 * 584];       // scores then probs (f16)
  __shared__ h16 lVT[2][256 * 40];   // v tiles transposed [c][k]
  __shared__ float lRed[32 * 8];

  {  // S = q @ k^T * 1/sqrt(256); 2 M-blocks x 4 key-groups of 144
    const int mblk = wid & 1, ng = wid >> 1;
    v8f acc[9];
#pragma unroll
    for (int j = 0; j < 9; ++j) acc[j] = vzero8();
    const h16* qrow = q + (size_t)(b * 4096 + q0 + 16 * mblk) * 256;
    for (int c0 = 0; c0 < 256; c0 += 32) {
      v16h a = a_frag(qrow + c0, 256, lane);
      v16h bf[9];
#pragma unroll
      for (int j = 0; j < 9; ++j) {
        int key = 144 * ng + 16 * j + (lane & 15);
        const h16* p =
            kmat + (size_t)(b * 576 + key) * 256 + c0 + 16 * (lane >> 4);
        bf[j] = make_frag(p, p + 8);
      }
#pragma unroll
      for (int j = 0; j < 9; ++j) acc[j] = wmma_f16(a, bf[j], acc[j]);
    }
#pragma unroll
    for (int j = 0; j < 9; ++j)
#pragma unroll
      for (int r = 0; r < 8; ++r) {
        int row = 16 * mblk + r + 8 * (lane >> 4);
        int key = 144 * ng + 16 * j + (lane & 15);
        lP[row * 584 + key] = (h16)(acc[j][r] * 0.0625f);
      }
  }
  __syncthreads();

  {  // softmax: 8 lanes per row
    int row = tid >> 3, sub = tid & 7;
    float m = -3.0e38f;
    for (int j2 = sub; j2 < 576; j2 += 8)
      m = fmaxf(m, (float)lP[row * 584 + j2]);
    lRed[row * 8 + sub] = m;
    __syncthreads();
    float rm = lRed[row * 8];
    for (int i = 1; i < 8; ++i) rm = fmaxf(rm, lRed[row * 8 + i]);
    float s = 0.f;
    for (int j2 = sub; j2 < 576; j2 += 8) {
      float e = expf((float)lP[row * 584 + j2] - rm);
      lP[row * 584 + j2] = (h16)e;
      s += e;
    }
    __syncthreads();
    lRed[row * 8 + sub] = s;
    __syncthreads();
    float rs = 0.f;
    for (int i = 0; i < 8; ++i) rs += lRed[row * 8 + i];
    float inv = 1.f / rs;
    for (int j2 = sub; j2 < 576; j2 += 8)
      lP[row * 584 + j2] = (h16)((float)lP[row * 584 + j2] * inv);
  }
  __syncthreads();

  {  // O = P @ V; 2 M-blocks x 4 col-groups of 64, double-buffered V tiles
    const int mblk = wid & 1, nq = wid >> 1;
    v8f acc[4];
#pragma unroll
    for (int j = 0; j < 4; ++j) acc[j] = vzero8();
    stage_vt(lVT[0], vmat, b, 0, tid);
    int buf = 0;
    for (int k0 = 0; k0 < 576; k0 += 32) {
      __syncthreads();
      if (k0 + 32 < 576) stage_vt(lVT[buf ^ 1], vmat, b, k0 + 32, tid);
      v16h a = a_frag(&lP[(16 * mblk) * 584 + k0], 584, lane);
      v16h bf[4];
#pragma unroll
      for (int j = 0; j < 4; ++j)
        bf[j] = b_frag(&lVT[buf][(64 * nq + 16 * j) * 40], 40, lane);
#pragma unroll
      for (int j = 0; j < 4; ++j) acc[j] = wmma_f16(a, bf[j], acc[j]);
      buf ^= 1;
    }
#pragma unroll
    for (int j = 0; j < 4; ++j)
#pragma unroll
      for (int r = 0; r < 8; ++r) {
        int row = q0 + 16 * mblk + r + 8 * (lane >> 4);
        int col = 64 * nq + 16 * j + (lane & 15);
        fused[((size_t)b * 4096 + row) * 256 + col] = (h16)acc[j][r];
      }
  }
}

// ---------------------------------------------------------------------------
// Launch
// ---------------------------------------------------------------------------
extern "C" void kernel_launch(void* const* d_in, const int* in_sizes, int n_in,
                              void* d_out, int out_size, void* d_ws,
                              size_t ws_size, hipStream_t stream) {
  (void)in_sizes; (void)n_in; (void)out_size; (void)ws_size;
  const float* seg_token    = (const float*)d_in[0];
  const float* depth_token  = (const float*)d_in[1];
  const float* refer_coords = (const float*)d_in[2];
  // d_in[3] = coarse_coords (unused)
  const float* global_pos   = (const float*)d_in[4];
  const float* dp1_w = (const float*)d_in[5];
  const float* dp1_b = (const float*)d_in[6];
  const float* dp2_w = (const float*)d_in[7];
  const float* dp2_b = (const float*)d_in[8];
  const float* kv1_w = (const float*)d_in[9];
  const float* kv1_b = (const float*)d_in[10];
  const float* kv2_w = (const float*)d_in[11];
  const float* kv2_b = (const float*)d_in[12];
  const float* sp1_w = (const float*)d_in[13];
  const float* sp1_b = (const float*)d_in[14];
  const float* sp2_w = (const float*)d_in[15];
  const float* sp2_b = (const float*)d_in[16];
  const float* q1_w  = (const float*)d_in[17];
  const float* q1_b  = (const float*)d_in[18];
  const float* q2_w  = (const float*)d_in[19];
  const float* q2_b  = (const float*)d_in[20];
  const float* f_w   = (const float*)d_in[21];
  const float* f_b   = (const float*)d_in[22];

  char* ws = (char*)d_ws;
  const size_t KV   = (size_t)8 * 576 * 256 * sizeof(h16);    // 2.36 MB each
  const size_t GEMO = (size_t)8 * 4096 * 1024 * sizeof(h16);  // 64 MB
  h16* kbuf  = (h16*)ws;
  h16* vbuf  = (h16*)(ws + KV);
  h16* gemo  = (h16*)(ws + 2 * KV);
  h16* h1    = (h16*)(ws + 2 * KV + GEMO);
  h16* qbuf  = gemo;  // reuse: gemo dead after q-MLP layer 1
  h16* fused = h1;    // reuse: h1 dead after q-MLP layer 2

  dim3 blk(256);
  refer_kv_kernel<<<dim3(9, 8), blk, 0, stream>>>(
      depth_token, global_pos, refer_coords, dp1_w, dp1_b, dp2_w, dp2_b,
      kv1_w, kv1_b, kv2_w, kv2_b, kbuf, vbuf);
  segrel_kernel<<<dim3(64, 8), blk, 0, stream>>>(seg_token, sp1_w, sp1_b,
                                                 sp2_w, sp2_b, gemo);
  geom_kernel<<<dim3(128), blk, 0, stream>>>(refer_coords, gemo);
  // q MLP layer 1: [32768,1024] @ [1024,1024], GELU
  gemm_x_w<1, 0><<<dim3(512, 8), blk, 0, stream>>>(gemo, 1024, q1_w, 1024,
                                                   q1_b, 1024, h1, 1024, nullptr);
  // q MLP layer 2: [32768,1024] @ [1024,256]
  gemm_x_w<0, 0><<<dim3(512, 2), blk, 0, stream>>>(h1, 1024, q2_w, 256, q2_b,
                                                   1024, qbuf, 256, nullptr);
  attn_kernel<<<dim3(128, 8), blk, 0, stream>>>(qbuf, kbuf, vbuf, fused);
  // final linear + transposed f32 store to (B, C, H, W)
  gemm_x_w<0, 1><<<dim3(512, 2), blk, 0, stream>>>(fused, 256, f_w, 256, f_b,
                                                   256, nullptr, 0,
                                                   (float*)d_out);
}
